// LinearRetriever_79766132621481
// MI455X (gfx1250) — compile-verified, matrix operand order checked by reading, not run on hardware
//
#include <hip/hip_runtime.h>

// ---------------------------------------------------------------------------
// MISA linear retriever for MI455X (gfx1250, wave32, WMMA).
// Phase 0: WH = f16(W) zero-padded to [224][512]
// Phase 1: xe = f16(x @ W^T + b) [B*L, 224], ye likewise [B*T, 224]
//          (col 200 = mask sentinel, cols 201..223 = 0)  -- WMMA f16/f32-acc
// Phase 2: per (i,j): colmax_t over l of xe[i]*ye[j]^T, then y-mask weighted
//          mean over t.  WMMA f16/f32-acc; A tiles double-buffered in LDS via
//          GLOBAL_LOAD_ASYNC_TO_LDS_B128 (ASYNCcnt) when the builtin exists,
//          else via a register-staged copy.
// ---------------------------------------------------------------------------

typedef _Float16 v8h  __attribute__((ext_vector_type(8)));
typedef _Float16 v16h __attribute__((ext_vector_type(16)));
typedef float    v8f  __attribute__((ext_vector_type(8)));
typedef int      v4i  __attribute__((ext_vector_type(4)));

#if __has_builtin(__builtin_amdgcn_global_load_async_to_lds_b128) && \
    __has_builtin(__builtin_amdgcn_s_wait_asynccnt)
#define USE_ASYNC_LDS 1
#else
#define USE_ASYNC_LDS 0
#endif

#if USE_ASYNC_LDS
// Signature (from hipcc diagnostics): param0 = v4i addrspace(1)* (global src),
// then LDS dst, imm offset, imm cpol; 4 args total.
#define GLD_ASYNC_B128(gsrc, ldst)                                        \
  __builtin_amdgcn_global_load_async_to_lds_b128(                         \
      (__attribute__((address_space(1))) v4i*)(void*)(gsrc),              \
      (__attribute__((address_space(3))) v4i*)(void*)(ldst), 0, 0)
#endif

namespace {
constexpr int kB  = 48;
constexpr int kL  = 512;
constexpr int kT  = 128;
constexpr int kD  = 512;   // input feature dim
constexpr int kE  = 200;   // embedding dim
constexpr int kEP = 224;   // padded to 7*32; col kE holds the mask sentinel
constexpr int kKC = kEP / 32;        // 7 K-chunks in the score GEMM
constexpr int kNT = kEP / 16;        // 14 n-tiles in projection
constexpr float kSent = 30000.0f;    // f16-safe sentinel magnitude
}

__device__ __forceinline__ v16h cat16(v8h lo, v8h hi) {
  return __builtin_shufflevector(lo, hi, 0, 1, 2, 3, 4, 5, 6, 7,
                                          8, 9, 10, 11, 12, 13, 14, 15);
}

__device__ __forceinline__ v8f wmma_f16(v16h a, v16h b, v8f c) {
  // (neg_a, A, neg_b, B, c_mod, C, reuse_a, reuse_b)
  return __builtin_amdgcn_wmma_f32_16x16x32_f16(false, a, false, b,
                                                (short)0, c, false, false);
}

__device__ __forceinline__ v8h cvt8(float4 a, float4 b) {
  v8h r;
  r[0] = (_Float16)a.x; r[1] = (_Float16)a.y;
  r[2] = (_Float16)a.z; r[3] = (_Float16)a.w;
  r[4] = (_Float16)b.x; r[5] = (_Float16)b.y;
  r[6] = (_Float16)b.z; r[7] = (_Float16)b.w;
  return r;
}

// ---------------------------------------------------------------------------
// Phase 0: WH[224][512] = f16(W[200][512]), rows 200..223 = 0.
// ---------------------------------------------------------------------------
__global__ __launch_bounds__(256)
void convert_w_kernel(const float* __restrict__ W, _Float16* __restrict__ WH) {
  const int idx = blockIdx.x * 256 + threadIdx.x;   // over 224*512
  const int row = idx >> 9;
  const float v = (row < kE) ? W[idx] : 0.0f;
  WH[idx] = (_Float16)v;
}

// ---------------------------------------------------------------------------
// Projection: OUT[R][224](f16) = X[R][512](f32) @ WH[224][512]^T + bias.
// One 16x16 tile per wave, K=512 (16 wmma). A converted f32->f16 on the fly;
// B read directly as f16 from padded WH.
// use_mask!=0 (x path): col 200 = mask[row]>0 ? 0 : -sent;  else col 200 = +sent.
// ---------------------------------------------------------------------------
__global__ __launch_bounds__(256)
void proj_kernel(const float* __restrict__ X, const _Float16* __restrict__ WH,
                 const float* __restrict__ bias, const float* __restrict__ mask,
                 _Float16* __restrict__ OUT, int R, int use_mask) {
  const int wave = threadIdx.x >> 5;
  const int lane = threadIdx.x & 31;
  const int wg   = blockIdx.x * 8 + wave;
  const int total = (R >> 4) * kNT;
  if (wg >= total) return;                    // wave-uniform exit (EXEC stays all-1)

  const int mt   = wg / kNT;
  const int nt   = wg % kNT;
  const int lrow = lane & 15;
  const int koff = (lane >> 4) << 3;          // 0 or 8 (16-bit A/B fragment layout)
  const int arow = mt * 16 + lrow;
  const int erow = nt * 16 + lrow;            // output embedding column (B frag row)

  const float*    xr  = X  + (size_t)arow * kD;
  const _Float16* wrh = WH + (size_t)erow * kD;   // rows 0..223 all valid in WH

  v8f acc = {0.f, 0.f, 0.f, 0.f, 0.f, 0.f, 0.f, 0.f};
#pragma unroll 4
  for (int kc = 0; kc < kD / 32; ++kc) {
    const int kb = kc * 32;
    float4 a0 = *(const float4*)(xr + kb + koff);
    float4 a1 = *(const float4*)(xr + kb + koff + 4);
    float4 a2 = *(const float4*)(xr + kb + 16 + koff);
    float4 a3 = *(const float4*)(xr + kb + 16 + koff + 4);
    v8h blo = *(const v8h*)(wrh + kb + koff);
    v8h bhi = *(const v8h*)(wrh + kb + 16 + koff);
    acc = wmma_f16(cat16(cvt8(a0, a1), cvt8(a2, a3)), cat16(blo, bhi), acc);
  }

  // Epilogue: C layout -> (row = mt*16 + r + (lane>=16)*8, e = nt*16 + lane&15)
  const int e     = erow;
  const int rbase = (lane >> 4) << 3;
  const float be  = (e < kE) ? bias[e] : 0.0f;
#pragma unroll
  for (int r = 0; r < 8; ++r) {
    const int row = mt * 16 + r + rbase;
    const float mq = mask[row];               // unconditional; uniform & cached
    const float sentv = use_mask ? ((mq > 0.0f) ? 0.0f : -kSent) : kSent;
    float v = acc[r] + be;
    if (e == kE)      v = sentv;
    else if (e > kE)  v = 0.0f;
    OUT[(size_t)row * kEP + e] = (_Float16)v;
  }
}

// ---------------------------------------------------------------------------
// Score + MISA reduction: one block per (i,j). Wave w owns t-tile w (B frags in
// VGPRs for all 32 m-tiles); A m-tiles of xe[i] double-buffered through LDS.
// ---------------------------------------------------------------------------
__global__ __launch_bounds__(256)
void score_kernel(const _Float16* __restrict__ XE,   // [48][512][224]
                  const _Float16* __restrict__ YE,   // [48][128][224]
                  const float* __restrict__ ymask,   // [48][128]
                  float* __restrict__ out) {         // [48][48]
  __shared__ __align__(16) _Float16 XA[2][16][232];  // 232-half stride: bank pad
  __shared__ float rnum[8], rden[8];

  const int i    = blockIdx.x;
  const int j    = blockIdx.y;
  const int tid  = threadIdx.x;
  const int wave = tid >> 5;
  const int lane = tid & 31;
  const int lrow = lane & 15;
  const int koff = (lane >> 4) << 3;

  // --- B fragments: ye[j], t-tile = wave, all 7 K-chunks resident in VGPRs ---
  v16h Bf[kKC];
  {
    const _Float16* yr = YE + ((size_t)j * kT + wave * 16 + lrow) * kEP;
#pragma unroll
    for (int kc = 0; kc < kKC; ++kc) {
      const int kb = kc * 32;
      v8h lo = *(const v8h*)(yr + kb + koff);
      v8h hi = *(const v8h*)(yr + kb + 16 + koff);
      Bf[kc] = cat16(lo, hi);
    }
  }

  // Cooperative-copy chunk coordinates (constant per thread; hoisted).
  const int c0 = tid, c1 = tid + 256;
  const int r0 = c0 / 28, q0 = c0 % 28;
  const int r1 = c1 / 28, q1 = c1 % 28;
  const bool has1 = (c1 < 448);              // 448 x 16B chunks per 16x224 tile

  const _Float16* xbase = XE + (size_t)i * kL * kEP;

  // --- stage A m-tile 0 into LDS buf 0 ---
#if USE_ASYNC_LDS
  GLD_ASYNC_B128(xbase + (size_t)r0 * kEP + q0 * 8, &XA[0][r0][q0 * 8]);
  if (has1)
    GLD_ASYNC_B128(xbase + (size_t)r1 * kEP + q1 * 8, &XA[0][r1][q1 * 8]);
  __builtin_amdgcn_s_wait_asynccnt(0);
#else
  *(uint4*)(&XA[0][r0][q0 * 8]) =
      *(const uint4*)(xbase + (size_t)r0 * kEP + q0 * 8);
  if (has1)
    *(uint4*)(&XA[0][r1][q1 * 8]) =
        *(const uint4*)(xbase + (size_t)r1 * kEP + q1 * 8);
#endif
  __syncthreads();

  float cmax[8];
#pragma unroll
  for (int r = 0; r < 8; ++r) cmax[r] = -3.0e38f;

  for (int mt = 0; mt < kL / 16; ++mt) {
    const int cur = mt & 1;
    const bool have = (mt + 1 < kL / 16);

#if USE_ASYNC_LDS
    // kick async DMA of the next m-tile into the other buffer
    if (have) {
      GLD_ASYNC_B128(xbase + (size_t)((mt + 1) * 16 + r0) * kEP + q0 * 8,
                     &XA[1 - cur][r0][q0 * 8]);
      if (has1)
        GLD_ASYNC_B128(xbase + (size_t)((mt + 1) * 16 + r1) * kEP + q1 * 8,
                       &XA[1 - cur][r1][q1 * 8]);
    }
#else
    uint4 p0, p1;
    if (have) {
      p0 = *(const uint4*)(xbase + (size_t)((mt + 1) * 16 + r0) * kEP + q0 * 8);
      if (has1)
        p1 = *(const uint4*)(xbase + (size_t)((mt + 1) * 16 + r1) * kEP + q1 * 8);
    }
#endif
    if (mt + 2 < kL / 16)  // warm caches for mt+2 (global_prefetch_b8)
      __builtin_prefetch(
          (const void*)(xbase + (size_t)((mt + 2) * 16 + r0) * kEP + q0 * 8), 0, 0);

    // compute: 7 WMMAs against the resident B fragments
    v8f acc = {0.f, 0.f, 0.f, 0.f, 0.f, 0.f, 0.f, 0.f};
    const _Float16* ar = &XA[cur][lrow][0];
#pragma unroll
    for (int kc = 0; kc < kKC; ++kc) {
      const int kb = kc * 32;
      v8h lo = *(const v8h*)(ar + kb + koff);
      v8h hi = *(const v8h*)(ar + kb + 16 + koff);
      acc = wmma_f16(cat16(lo, hi), Bf[kc], acc);
    }
#pragma unroll
    for (int r = 0; r < 8; ++r) cmax[r] = fmaxf(cmax[r], acc[r]);

#if USE_ASYNC_LDS
    __builtin_amdgcn_s_wait_asynccnt(0);   // my fills of buf 1-cur landed
#else
    if (have) {
      *(uint4*)(&XA[1 - cur][r0][q0 * 8]) = p0;
      if (has1) *(uint4*)(&XA[1 - cur][r1][q1 * 8]) = p1;
    }
#endif
    __syncthreads();                       // everyone's fills visible
  }

  // --- reduction: per-lane max over VGPRs, merge M-halves via lane-xor-16 ---
  float m = cmax[0];
#pragma unroll
  for (int r = 1; r < 8; ++r) m = fmaxf(m, cmax[r]);
  m = fmaxf(m, __shfl_xor(m, 16, 32));   // lanes n and n+16 hold column n

  const int t   = wave * 16 + lrow;
  const float w = ymask[(size_t)j * kT + t];
  float num = m * w;
  float den = w;
#pragma unroll
  for (int off = 8; off >= 1; off >>= 1) {   // sum within each 16-lane half
    num += __shfl_xor(num, off, 32);
    den += __shfl_xor(den, off, 32);
  }
  if (lane == 0) { rnum[wave] = num; rden[wave] = den; }
  __syncthreads();
  if (tid == 0) {
    float N = 0.f, Ts = 0.f;
#pragma unroll
    for (int wv = 0; wv < 8; ++wv) { N += rnum[wv]; Ts += rden[wv]; }
    out[i * kB + j] = (Ts > 0.0f) ? N / fmaxf(Ts, 1.0f) : 0.0f;
  }
}

// ---------------------------------------------------------------------------
extern "C" void kernel_launch(void* const* d_in, const int* in_sizes, int n_in,
                              void* d_out, int out_size, void* d_ws,
                              size_t ws_size, hipStream_t stream) {
  (void)in_sizes; (void)n_in; (void)out_size; (void)ws_size;
  const float* x  = (const float*)d_in[0];   // [48,512,512]
  const float* y  = (const float*)d_in[1];   // [48,128,512]
  const float* xm = (const float*)d_in[2];   // [48,512]
  const float* ym = (const float*)d_in[3];   // [48,128]
  const float* W  = (const float*)d_in[4];   // [200,512]
  const float* b  = (const float*)d_in[5];   // [200]
  float* out = (float*)d_out;                // [48,48]

  _Float16* XE = (_Float16*)d_ws;                          // 48*512*224 f16
  _Float16* YE = XE + (size_t)kB * kL * kEP;               // 48*128*224 f16
  _Float16* WH = YE + (size_t)kB * kT * kEP;               // 224*512 f16

  const int Rx = kB * kL;                                  // 24576
  const int Ry = kB * kT;                                  // 6144
  const int wavesX = (Rx / 16) * kNT;                      // 21504
  const int wavesY = (Ry / 16) * kNT;                      // 5376

  convert_w_kernel<<<dim3((kEP * kD) / 256), 256, 0, stream>>>(W, WH);
  proj_kernel<<<dim3((wavesX + 7) / 8), 256, 0, stream>>>(x, WH, b, xm, XE, Rx, 1);
  proj_kernel<<<dim3((wavesY + 7) / 8), 256, 0, stream>>>(y, WH, b, ym, YE, Ry, 0);
  score_kernel<<<dim3(kB, kB), 256, 0, stream>>>(XE, YE, ym, out);
}
